// SparseSO3Conv_40561671143689
// MI455X (gfx1250) — matrix operand order repaired and checked by reading, not compiled, not run on hardware
//
#include <hip/hip_runtime.h>
#include <hip/hip_bf16.h>

typedef __bf16 bf16;
typedef __attribute__((ext_vector_type(16))) __bf16 v16bf;
typedef __attribute__((ext_vector_type(8)))  __bf16 v8bf;
typedef __attribute__((ext_vector_type(8)))  float  v8f;
typedef __attribute__((ext_vector_type(4)))  unsigned int v4u;
typedef __attribute__((ext_vector_type(8)))  int v8i_;
typedef __attribute__((ext_vector_type(4)))  int v4i_;

#define BN    4
#define NN    4096
#define KK    32
#define NINF  64
#define RANK  16
#define NOUTF 128
#define HIDF  64
#define GN    16      // n-values per block
#define NTHREADS 256

// ---- LDS layout (bytes), phase-overlapped unions -------------------------
// region1 [0,10240): Phase A: pts f32[16][32][3] (6144) + rmean f32[16][3]
//                    Phase E: wocT bf16[128][40] (10240, padded rows)
//                    LN:      part f32[8][16][2] (1024) + rowstat f32[16][2]
static constexpr int PTS_OFF     = 0;
static constexpr int RMEAN_OFF   = 6144;
static constexpr int WOCT_OFF    = 0;
static constexpr int PART_OFF    = 0;
static constexpr int ROWSTAT_OFF = 1024;
static constexpr int W1_OFF      = 10240;   // f32[8][64]        2048
static constexpr int B1_OFF      = 12288;   // f32[64]            256
static constexpr int G1_OFF      = 12544;   // f32[64]            256
static constexpr int BE1_OFF     = 12800;   // f32[64]            256
static constexpr int B2_OFF      = 13056;   // f32[16]             64
static constexpr int W2T_OFF     = 13120;   // bf16[16][64]      2048
static constexpr int KERNT_OFF   = 15168;   // bf16[16][16][32] 16384  [n][r][k]
static constexpr int CONT_OFF    = 31552;   // bf16[16][1032]   33024 (padded)
                                            //  union: hdn bf16[256][64] 32768
static constexpr int SMEM_BYTES  = 64576;

static constexpr int CONT_STRIDE = 1032;    // 2064 B rows: 16B aligned, bank step 4
static constexpr int WOCT_STRIDE = 40;      // 80 B rows:   16B aligned, bank step 20

#define SHUF16(a, b) __builtin_shufflevector(a, b, 0,1,2,3,4,5,6,7,8,9,10,11,12,13,14,15)

__global__ __launch_bounds__(NTHREADS)
void so3conv_fused_wmma_kernel(
    const float* __restrict__ fp,    // feat_points (B,N,K,3)
    const float* __restrict__ feat,  // feat        (B,N,K,NIN)
    const float* __restrict__ ep,    // eval_points (B,N,3)
    const float* __restrict__ w1, const float* __restrict__ b1,
    const float* __restrict__ g1, const float* __restrict__ be1,
    const float* __restrict__ w2, const float* __restrict__ b2,
    const float* __restrict__ wo, const float* __restrict__ bwo,
    const float* __restrict__ gln, const float* __restrict__ bln,
    float* __restrict__ out)
{
  __shared__ __align__(128) char smem[SMEM_BYTES];

  float* ptsS   = (float*)(smem + PTS_OFF);
  float* rmeanS = (float*)(smem + RMEAN_OFF);
  float* w1S    = (float*)(smem + W1_OFF);
  float* b1S    = (float*)(smem + B1_OFF);
  float* g1S    = (float*)(smem + G1_OFF);
  float* be1S   = (float*)(smem + BE1_OFF);
  float* b2S    = (float*)(smem + B2_OFF);
  bf16*  w2T    = (bf16*)(smem + W2T_OFF);
  bf16*  kernT  = (bf16*)(smem + KERNT_OFF);
  bf16*  contS  = (bf16*)(smem + CONT_OFF);
  bf16*  hdnS   = (bf16*)(smem + CONT_OFF);   // union with contS

  const int t      = threadIdx.x;
  const int bb     = blockIdx.x;
  const int b      = bb >> 8;             // NN/GN = 256 blocks per batch
  const int n_base = (bb & 255) << 4;
  const size_t bn0 = (size_t)b * NN + n_base;

  const int lane = t & 31, wv = t >> 5;
  const int g = lane >> 4, l16 = lane & 15;

  // ---------------- Stage point tile via Tensor Data Mover ----------------
  // Contiguous (16,32,3) f32 slice = 1536 elements -> LDS[PTS_OFF].
  if (wv == 0) {
    const unsigned long long ga =
        (unsigned long long)(const void*)(fp + bn0 * (KK * 3));
    v4u g0;
    g0[0] = 1u;                                   // count=1 (valid user D#)
    g0[1] = (unsigned)PTS_OFF;                    // lds_addr (bytes)
    g0[2] = (unsigned)(ga & 0xffffffffull);       // global_addr[31:0]
    g0[3] = (unsigned)((ga >> 32) & 0x1ffffffull) // global_addr[56:32]
          | (2u << 30);                           // type=2 ("image")
    v8i_ g1v = {};
    g1v[0] = (2 << 16);                           // data_size=2 -> 4 bytes
    g1v[1] = (1536 << 16);                        // tensor_dim0[15:0] in [63:48]
    g1v[2] = (1 << 16);                           // tensor_dim0 hi=0, tensor_dim1=1
    g1v[3] = (1536 << 16);                        // tile_dim0 = 1536
    g1v[4] = 0;                                   // tile_dim1/2 unused
    g1v[5] = 1536;                                // tensor_dim0_stride lo
    g1v[6] = 0; g1v[7] = 0;
    v4i_ gz4 = {};
    v8i_ gz8 = {};
    // 6-arg form (clang-23 / therock headers): (g0, g1, g2, g3, g4, cpol)
    __builtin_amdgcn_tensor_load_to_lds(g0, g1v, gz4, gz4, gz8, 0);
    __builtin_amdgcn_s_wait_tensorcnt(0);
  }

  // ---------------- Stage weights (w2 transposed -> bf16) -----------------
  for (int i = t; i < 8 * HIDF;  i += NTHREADS) w1S[i] = w1[i];
  for (int i = t; i < HIDF;      i += NTHREADS) { b1S[i] = b1[i]; g1S[i] = g1[i]; be1S[i] = be1[i]; }
  if (t < RANK) b2S[t] = b2[t];
  for (int i = t; i < HIDF*RANK; i += NTHREADS) {
    const int r = i >> 6, j = i & 63;             // w2T[r][j] = w2[j][r]
    w2T[i] = (bf16)w2[j * RANK + r];
  }
  __syncthreads();

  // ---------------- Per-n neighbor centroid -------------------------------
  if (t < GN * 3) {
    const int n = t / 3, d = t % 3;
    float s = 0.f;
    for (int k = 0; k < KK; ++k) s += ptsS[(n * KK + k) * 3 + d];
    rmeanS[n * 3 + d] = s * (1.f / KK);
  }
  __syncthreads();

  // ======== Phases A-C in two 256-row passes ==============================
  for (int pass = 0; pass < 2; ++pass) {
    // ---- Phase A/B (scalar): rifeat + (8->64) + LN + ReLU -> hdnS[t][64]
    {
      const int row = t + pass * NTHREADS;        // 512 rows = 16 n * 32 k
      const int n = row >> 5, k = row & 31;
      const float prx = ptsS[(n * KK + k) * 3 + 0];
      const float pry = ptsS[(n * KK + k) * 3 + 1];
      const float prz = ptsS[(n * KK + k) * 3 + 2];
      const float rmx = rmeanS[n*3+0], rmy = rmeanS[n*3+1], rmz = rmeanS[n*3+2];
      const size_t ng = bn0 + n;
      const float psx = ep[ng*3+0], psy = ep[ng*3+1], psz = ep[ng*3+2];

      const float l1x = rmx - prx, l1y = rmy - pry, l1z = rmz - prz;
      const float l2x = prx - psx, l2y = pry - psy, l2z = prz - psz;
      const float l3x = psx - rmx, l3y = psy - rmy, l3z = psz - rmz;
      const float l1n = sqrtf(l1x*l1x + l1y*l1y + l1z*l1z);
      const float l2n = sqrtf(l2x*l2x + l2y*l2y + l2z*l2z);
      const float l3n = sqrtf(l3x*l3x + l3y*l3y + l3z*l3z);
      const float th1 = (l1x*l2x + l1y*l2y + l1z*l2z) / (l1n*l2n + 1e-7f);
      const float th2 = (l2x*l3x + l2y*l3y + l2z*l3z) / (l2n*l3n + 1e-7f);
      const float th3 = (l3x*l1x + l3y*l1y + l3z*l1z) / (l3n*l1n + 1e-7f);

      const float az  = atan2f(psy, psx);
      const float psn = sqrtf(psx*psx + psy*psy + psz*psz);
      const float pol = acosf(fminf(fmaxf(psz / psn, -1.f), 1.f));
      // row 2 of Zm(-psz) @ Ym(-pol) @ Zm(-az) == (-s2*c3, s2*s3, c2)
      const float a2 = -pol, a3 = -az;
      const float s2 = sinf(a2), c2 = cosf(a2), s3 = sinf(a3), c3 = cosf(a3);
      const float m20 = -s2 * c3, m21 = s2 * s3, m22 = c2;

      const float h    = sqrtf(prx*prx + pry*pry + prz*prz);
      const float res2 = (m20*prx + m21*pry + m22*prz) / h;
      const float txj  = acosf(fminf(fmaxf(res2, -1.f), 1.f)) * 0.3183098861837907f;

      float ri[8] = { txj, h, l1n, l2n, l3n, th1, th2, th3 };

      float hid[HIDF];
      float s = 0.f;
      #pragma unroll
      for (int j = 0; j < HIDF; ++j) {
        float a = b1S[j];
        #pragma unroll
        for (int f = 0; f < 8; ++f) a += ri[f] * w1S[f * HIDF + j];
        hid[j] = a; s += a;
      }
      const float mu = s * (1.f / HIDF);
      float q = 0.f;
      #pragma unroll
      for (int j = 0; j < HIDF; ++j) { const float d = hid[j] - mu; q += d * d; }
      const float rstd = rsqrtf(q * (1.f / HIDF) + 1e-5f);
      #pragma unroll
      for (int j = 0; j < HIDF; ++j) {
        const float xn = (hid[j] - mu) * rstd * g1S[j] + be1S[j];
        hdnS[t * HIDF + j] = (bf16)fmaxf(xn, 0.f);
      }
    }
    __syncthreads();

    // ---- Phase C (WMMA): kern(512x16) = hdn(512x64) @ w2(64x16) + b2 ----
    for (int sub = 0; sub < 2; ++sub) {
      const int rtp = wv * 2 + sub;               // row-tile in this pass
      v8f c = {};
      #pragma unroll
      for (int ks = 0; ks < 2; ++ks) {
        const v8bf* ap = (const v8bf*)(hdnS + ((rtp*16 + l16) * HIDF + ks*32 + 8*g));
        const v8bf a0 = ap[0], a1 = ap[2];        // j = ks*32+8g+e | +16
        const v8bf* bp = (const v8bf*)(w2T + (l16 * HIDF + ks*32 + 8*g));
        const v8bf b0 = bp[0], b1v = bp[2];       // j = ks*32+8g+e | +16
        const v16bf av = SHUF16(a0, a1);
        const v16bf bv = SHUF16(b0, b1v);
        c = __builtin_amdgcn_wmma_f32_16x16x32_bf16(false, av, false, bv,
                                                    (short)0, c, false, false);
      }
      const float bias = b2S[l16];                // N-col = r = l16
      #pragma unroll
      for (int v = 0; v < 8; ++v) {
        const int grow = pass * 256 + rtp * 16 + v + 8 * g;
        const int n = grow >> 5, k = grow & 31;
        kernT[(n * RANK + l16) * KK + k] = (bf16)(c[v] + bias);
      }
    }
    __syncthreads();
  }

  // ---------------- Phase D: contracted = kern^T(16x32) @ feat(32x64) -----
  for (int sub = 0; sub < 2; ++sub) {
    const int nl = wv * 2 + sub;                  // wave owns 2 of 16 n
    // A fragment from kernT[n][r][k]: contiguous k-runs -> 2x ds_load_b128
    const v8bf* kp = (const v8bf*)(kernT + ((nl * RANK + l16) * KK + 16 * g));
    const v8bf k0 = kp[0], k1 = kp[1];            // k = 16g+e
    const v16bf a = SHUF16(k0, k1);

    const float* fbase = feat + (bn0 + nl) * (size_t)(KK * NINF);
    #pragma unroll
    for (int tile = 0; tile < 4; ++tile) {
      v16bf bm;                                   // B 32x16: k = 16g+e
      #pragma unroll
      for (int e = 0; e < 16; ++e)
        bm[e] = (bf16)fbase[(16 * g + e) * NINF + tile * 16 + l16];
      v8f c = {};
      c = __builtin_amdgcn_wmma_f32_16x16x32_bf16(false, a, false, bm,
                                                  (short)0, c, false, false);
      #pragma unroll
      for (int v = 0; v < 8; ++v) {
        const int r = v + 8 * g;
        contS[nl * CONT_STRIDE + r * NINF + tile * 16 + l16] = (bf16)c[v];
      }
    }
  }
  __syncthreads();

  // ---------------- Phase E: out(16x128) = contracted(16x1024) @ wo -------
  bf16* wocT = (bf16*)(smem + WOCT_OFF);          // [o][kr], padded rows
  v8f acc = {};
  for (int c = 0; c < 32; ++c) {
    const float* wsrc = wo + (size_t)c * 32 * NOUTF;
    for (int i = t; i < 32 * NOUTF; i += NTHREADS) {
      const int kr = i >> 7, o = i & 127;         // coalesced global read
      wocT[o * WOCT_STRIDE + kr] = (bf16)wsrc[i];
    }
    if (c + 1 < 32)
      __builtin_prefetch(wo + (size_t)(c + 1) * 32 * NOUTF + t * 16, 0, 1);
    __syncthreads();

    // A: contracted rows (m = n_local = l16), contiguous kk runs
    const v8bf* cp = (const v8bf*)(contS + (l16 * CONT_STRIDE + c * 32 + 8 * g));
    const v8bf c0 = cp[0], c1 = cp[2];
    const v16bf av = SHUF16(c0, c1);
    // B: wave's 16-wide o-tile, contiguous kr runs
    const v8bf* wp = (const v8bf*)(wocT + ((wv * 16 + l16) * WOCT_STRIDE + 8 * g));
    const v8bf w0 = wp[0], w1v = wp[2];
    const v16bf bv = SHUF16(w0, w1v);
    acc = __builtin_amdgcn_wmma_f32_16x16x32_bf16(false, av, false, bv,
                                                  (short)0, acc, false, false);
    __syncthreads();
  }

  // ---------------- Bias + LayerNorm over NOUT=128 ------------------------
  const int o = wv * 16 + l16;
  const float bw = bwo[o], gl = gln[o], bl = bln[o];
  #pragma unroll
  for (int v = 0; v < 8; ++v) acc[v] += bw;

  float psum[8], psq[8];
  #pragma unroll
  for (int v = 0; v < 8; ++v) {
    float s = acc[v], q = acc[v] * acc[v];
    #pragma unroll
    for (int off = 1; off < 16; off <<= 1) {
      s += __shfl_xor(s, off, 16);
      q += __shfl_xor(q, off, 16);
    }
    psum[v] = s; psq[v] = q;
  }
  float* partS  = (float*)(smem + PART_OFF);      // aliases dead wocT buffer
  float* rstatS = (float*)(smem + ROWSTAT_OFF);
  if (l16 == 0) {
    #pragma unroll
    for (int v = 0; v < 8; ++v) {
      const int m = v + 8 * g;
      partS[(wv * 16 + m) * 2 + 0] = psum[v];
      partS[(wv * 16 + m) * 2 + 1] = psq[v];
    }
  }
  __syncthreads();
  if (t < 16) {
    float s = 0.f, q = 0.f;
    for (int w = 0; w < 8; ++w) { s += partS[(w*16 + t)*2]; q += partS[(w*16 + t)*2 + 1]; }
    const float mu  = s * (1.f / NOUTF);
    const float var = q * (1.f / NOUTF) - mu * mu;
    rstatS[t * 2 + 0] = mu;
    rstatS[t * 2 + 1] = rsqrtf(var + 1e-5f);
  }
  __syncthreads();
  #pragma unroll
  for (int v = 0; v < 8; ++v) {
    const int m = v + 8 * g;                      // row = n_local
    const float mu = rstatS[m * 2], rs = rstatS[m * 2 + 1];
    out[(bn0 + m) * NOUTF + o] = (acc[v] - mu) * rs * gl + bl;
  }
}

extern "C" void kernel_launch(void* const* d_in, const int* in_sizes, int n_in,
                              void* d_out, int out_size, void* d_ws, size_t ws_size,
                              hipStream_t stream) {
  (void)in_sizes; (void)n_in; (void)out_size; (void)d_ws; (void)ws_size;
  const float* fp   = (const float*)d_in[0];
  const float* feat = (const float*)d_in[1];
  const float* ep   = (const float*)d_in[2];
  const float* w1   = (const float*)d_in[3];
  const float* b1   = (const float*)d_in[4];
  const float* g1   = (const float*)d_in[5];
  const float* be1  = (const float*)d_in[6];
  const float* w2   = (const float*)d_in[7];
  const float* b2   = (const float*)d_in[8];
  const float* wo   = (const float*)d_in[9];
  const float* bwo  = (const float*)d_in[10];
  const float* gln  = (const float*)d_in[11];
  const float* bln  = (const float*)d_in[12];

  dim3 grid(BN * (NN / GN));   // 1024 blocks, each = 16 n-values
  so3conv_fused_wmma_kernel<<<grid, NTHREADS, 0, stream>>>(
      fp, feat, ep, w1, b1, g1, be1, w2, b2, wo, bwo, gln, bln, (float*)d_out);
}
